// AttentionConvInput_54125177864329
// MI455X (gfx1250) — compile-verified, matrix operand order checked by reading, not run on hardware
//
#include <hip/hip_runtime.h>
#include <hip/hip_bf16.h>

// Problem constants (reference: B, C, L, D = 32, 1, 1024, 128)
constexpr int B = 32;
constexpr int L = 1024;
constexpr int D = 128;

typedef __attribute__((ext_vector_type(16))) _Float16 v16h;
typedef __attribute__((ext_vector_type(8)))  _Float16 h8;
typedef __attribute__((ext_vector_type(8)))  float    v8f;
typedef __attribute__((ext_vector_type(4)))  int      v4i;

union V16 { v16h v; h8 h[2]; };

// --- CDNA5 async global->LDS copy (ASYNCcnt-tracked DMA) with fallback -----
#ifdef __has_builtin
#if __has_builtin(__builtin_amdgcn_global_load_async_to_lds_b128) && \
    __has_builtin(__builtin_amdgcn_s_wait_asynccnt)
#define HAVE_ASYNC_LDS 1
#endif
#endif
#ifndef HAVE_ASYNC_LDS
#define HAVE_ASYNC_LDS 0
#endif

__device__ __forceinline__ void async_cp16(const _Float16* g, _Float16* l) {
#if HAVE_ASYNC_LDS
  using gp_t = v4i __attribute__((address_space(1)))*;  // non-const per builtin
  using lp_t = v4i __attribute__((address_space(3)))*;
  // Flat shared address: low 32 bits are the LDS offset -> truncate.
  __builtin_amdgcn_global_load_async_to_lds_b128(
      (gp_t)(unsigned long long)g,
      (lp_t)(unsigned int)(unsigned long long)l, 0, 0);
#else
  *(v4i*)l = *(const v4i*)g;  // sync fallback: global_load_b128 + ds_store
#endif
}

#if HAVE_ASYNC_LDS
#define WAIT_ASYNC(n) __builtin_amdgcn_s_wait_asynccnt(n)
#else
#define WAIT_ASYNC(n) ((void)0)
#endif

// ---------------------------------------------------------------------------
// Prep 1: f32 -> f16 copies of x0/x1 into workspace, and write channel 0 of
// f0/f1 (the concat identity part).
// ---------------------------------------------------------------------------
__global__ void prep_convert(const float* __restrict__ x0,
                             const float* __restrict__ x1,
                             _Float16* __restrict__ x0h,
                             _Float16* __restrict__ x1h,
                             float* __restrict__ f0,
                             float* __restrict__ f1) {
  size_t i = (size_t)blockIdx.x * blockDim.x + threadIdx.x;  // over B*L*D
  size_t b   = i / ((size_t)L * D);
  size_t rem = i % ((size_t)L * D);
  float v0 = x0[i];
  float v1 = x1[i];
  x0h[i] = (_Float16)v0;
  x1h[i] = (_Float16)v1;
  f0[(b * 2) * (size_t)(L * D) + rem] = v0;
  f1[(b * 2) * (size_t)(L * D) + rem] = v1;
}

// ---------------------------------------------------------------------------
// Prep 2: per-row squared norms in f32 (one wave32 per row, float4 per lane,
// xor-shuffle reduction).
// ---------------------------------------------------------------------------
__global__ void prep_sq(const float* __restrict__ x0,
                        const float* __restrict__ x1,
                        float* __restrict__ sq0,
                        float* __restrict__ sq1) {
  const int lane = threadIdx.x & 31;
  const int w    = threadIdx.x >> 5;
  const int row  = blockIdx.x * 8 + w;  // B*L rows total, 8 waves/block
  const float4 a = ((const float4*)(x0 + (size_t)row * D))[lane];
  const float4 c = ((const float4*)(x1 + (size_t)row * D))[lane];
  float s0 = a.x * a.x + a.y * a.y + a.z * a.z + a.w * a.w;
  float s1 = c.x * c.x + c.y * c.y + c.z * c.z + c.w * c.w;
#pragma unroll
  for (int off = 16; off > 0; off >>= 1) {
    s0 += __shfl_xor(s0, off, 32);
    s1 += __shfl_xor(s1, off, 32);
  }
  if (lane == 0) {
    sq0[row] = s0;
    sq1[row] = s1;
  }
}

// ---------------------------------------------------------------------------
// Prep 3: W (L x D f32) -> Wt (D x L f16) so B-operand columns are contiguous.
// ---------------------------------------------------------------------------
__global__ void prep_wt(const float* __restrict__ W0,
                        const float* __restrict__ W1,
                        _Float16* __restrict__ W0t,
                        _Float16* __restrict__ W1t) {
  int i = blockIdx.x * blockDim.x + threadIdx.x;  // over L*D
  int j = i / D;
  int d = i % D;
  W0t[(size_t)d * L + j] = (_Float16)W0[i];
  W1t[(size_t)d * L + j] = (_Float16)W1[i];
}

// ---------------------------------------------------------------------------
// Fused kernel. grid = (L/128, B, 2); z==0: a0 = A@W0 -> f0 ch1 (rows = x0);
// z==1: a1 = A^T@W1 -> f1 ch1 (rows = x1; A^T[j,i] = f(dist(x1_j, x0_i))).
// Per 64-wide column chunk: async-DMA stage Xo chunk + Wt chunk into LDS
// (double-buffered, overlapped with compute via ASYNCcnt), 16 inner WMMAs,
// elementwise A = 1/(1+sqrt(max(.,0))) -> f16 via wave-private LDS transpose,
// 16 accumulate WMMAs against the staged Wt chunk. B-operand LDS loads are
// software-pipelined through a rotating 3-slot register buffer so each WMMA
// waits on a load issued ~3 WMMAs earlier (partial s_wait_dscnt).
// ---------------------------------------------------------------------------
__global__ __launch_bounds__(256) void attn_fused(
    const _Float16* __restrict__ x0h, const _Float16* __restrict__ x1h,
    const float* __restrict__ sq0, const float* __restrict__ sq1,
    const _Float16* __restrict__ W0t, const _Float16* __restrict__ W1t,
    float* __restrict__ f0, float* __restrict__ f1) {
  constexpr int TM = 128;      // rows per block
  constexpr int KC = 64;       // column-chunk width
  constexpr int NC = L / KC;   // 16 chunks
  __shared__ _Float16 At[TM * KC];       // 16 KB, wave-private 16-row strips
  __shared__ _Float16 XoS[2][KC * D];    // 2 x 16 KB staged Xo chunk
  __shared__ _Float16 WtS[2][D * KC];    // 2 x 16 KB staged Wt chunk

  const int tid  = threadIdx.x;
  const int lane = tid & 31;
  const int w    = tid >> 5;
  const int mr   = lane & 15;         // row/col within 16-tile
  const int hk   = (lane >> 4) << 3;  // K-base for 16-bit operand layout
  const int hm   = hk;                // M-offset in C-layout

  const int tile = blockIdx.x;
  const int b    = blockIdx.y;
  const int z    = blockIdx.z;

  const _Float16* Xm  = z ? x1h : x0h;  // "M-side" rows
  const _Float16* Xo  = z ? x0h : x1h;  // "other" rows (columns of A)
  const float*    sqm = z ? sq1 : sq0;
  const float*    sqo = z ? sq0 : sq1;
  const _Float16* Wt  = z ? W1t : W0t;
  float* out = (z ? f1 : f0) + ((size_t)b * 2 + 1) * (size_t)(L * D);  // ch 1

  const int mBase = tile * TM + w * 16;
  const _Float16* XoB = Xo + (size_t)b * L * D;  // batch base

  // --- hoisted A-operands for inner product: this wave's 16 rows, K = D
  const _Float16* XmRow = Xm + ((size_t)(b * L + mBase + mr)) * D;
  V16 Aop[4];
#pragma unroll
  for (int kk = 0; kk < 4; ++kk) {
    const _Float16* p = XmRow + kk * 32 + hk;
    Aop[kk].h[0] = *(const h8*)p;
    Aop[kk].h[1] = *(const h8*)(p + 16);
  }

  // sqa values matching the accumulator's C-layout (element e -> row e+hm)
  float sqa_v[8];
#pragma unroll
  for (int e = 0; e < 8; ++e) sqa_v[e] = sqm[b * L + mBase + e + hm];

  v8f R[8];
#pragma unroll
  for (int t = 0; t < 8; ++t) R[t] = (v8f)(0.0f);

  // --- stage chunk `c` into buffer `buf` (8 x b128 per thread: 4 Xo + 4 Wt)
  auto stage = [&](int c, int buf) {
    const _Float16* xoSrc = XoB + (size_t)(c * KC) * D;  // KC*D contiguous
    const _Float16* wtSrc = Wt + c * KC;
#pragma unroll
    for (int k = 0; k < 4; ++k) {
      int q = tid + 256 * k;  // 0..1023 16-byte packets
      async_cp16(xoSrc + q * 8, &XoS[buf][q * 8]);
    }
#pragma unroll
    for (int k = 0; k < 4; ++k) {
      int q  = tid + 256 * k;
      int dd = q >> 3;            // 0..127 (d row of Wt)
      int jj = (q & 7) * 8;       // 0..56  (halves within chunk row)
      async_cp16(wtSrc + (size_t)dd * L + jj, &WtS[buf][dd * KC + jj]);
    }
  };

  stage(0, 0);

  for (int c = 0; c < NC; ++c) {
    const int cur = c & 1;
    if (c + 1 < NC) {
      stage(c + 1, cur ^ 1);
      WAIT_ASYNC(8);   // chunk c's 8 per-wave async ops have landed
    } else {
      WAIT_ASYNC(0);
    }
    __syncthreads();   // all waves' staged data visible

    const int jc = c * KC;
    const _Float16* XoC = XoS[cur];
    const _Float16* WtC = WtS[cur];

    float sqb_v[4];
#pragma unroll
    for (int tn = 0; tn < 4; ++tn)
      sqb_v[tn] = sqo[b * L + jc + tn * 16 + mr];

    // --- inner = Xm_rows @ Xo_rows^T for this 128x64 chunk (per wave 16x64)
    // i = tn*4 + kk; B-operands pipelined via 3-slot rotating buffer.
    auto loadXo = [&](int i) -> V16 {
      const int tn = i >> 2, kk = i & 3;
      const _Float16* p = XoC + (tn * 16 + mr) * D + kk * 32 + hk;
      V16 t;
      t.h[0] = *(const h8*)p;
      t.h[1] = *(const h8*)(p + 16);
      return t;
    };
    v8f S[4];
#pragma unroll
    for (int tn = 0; tn < 4; ++tn) S[tn] = (v8f)(0.0f);
    V16 Bb[3];
#pragma unroll
    for (int i = 0; i < 3; ++i) Bb[i] = loadXo(i);
#pragma unroll
    for (int i = 0; i < 16; ++i) {
      S[i >> 2] = __builtin_amdgcn_wmma_f32_16x16x32_f16(
          false, Aop[i & 3].v, false, Bb[i % 3].v, (short)0, S[i >> 2], false,
          false);
      if (i + 3 < 16) Bb[i % 3] = loadXo(i + 3);
    }

    // --- elementwise A = 1/(1+sqrt(max(sqa+sqb-2*inner,0))), f16 into LDS.
    // Wave w writes rows [w*16, w*16+16) and reads only those rows back:
    // wave-private, DScnt orders the intra-wave store->load hazard.
#pragma unroll
    for (int tn = 0; tn < 4; ++tn) {
      const int nl = tn * 16 + mr;
#pragma unroll
      for (int e = 0; e < 8; ++e) {
        const int ml = w * 16 + e + hm;
        float d2 = fmaxf(sqa_v[e] + sqb_v[tn] - 2.0f * S[tn][e], 0.0f);
        float av = 1.0f / (1.0f + sqrtf(d2));
        At[ml * KC + nl] = (_Float16)av;
      }
    }

    // --- R += A_tile(16 x KC) @ Wt_chunk(KC x 128), B from staged LDS.
    // i = kk*8 + tn; same 3-slot pipelined B loads; both Aop2 hoisted.
    auto loadWt = [&](int i) -> V16 {
      const int kk = i >> 3, tn = i & 7;
      const _Float16* p = WtC + (tn * 16 + mr) * KC + kk * 32 + hk;
      V16 t;
      t.h[0] = *(const h8*)p;
      t.h[1] = *(const h8*)(p + 16);
      return t;
    };
    V16 Aop2[2];
#pragma unroll
    for (int kk = 0; kk < 2; ++kk) {
      const _Float16* ap = At + (w * 16 + mr) * KC + kk * 32 + hk;
      Aop2[kk].h[0] = *(const h8*)ap;
      Aop2[kk].h[1] = *(const h8*)(ap + 16);
    }
    V16 Wb[3];
#pragma unroll
    for (int i = 0; i < 3; ++i) Wb[i] = loadWt(i);
#pragma unroll
    for (int i = 0; i < 16; ++i) {
      R[i & 7] = __builtin_amdgcn_wmma_f32_16x16x32_f16(
          false, Aop2[i >> 3].v, false, Wb[i % 3].v, (short)0, R[i & 7], false,
          false);
      if (i + 3 < 16) Wb[i % 3] = loadWt(i + 3);
    }

    __syncthreads();  // reads of buf[cur] done before it is re-staged
  }

  // --- write 16x128 f32 result strip (C-layout scatter, lanes coalesce in n)
#pragma unroll
  for (int tn = 0; tn < 8; ++tn) {
    const int n = tn * 16 + mr;
#pragma unroll
    for (int e = 0; e < 8; ++e) {
      const int m = mBase + e + hm;
      out[(size_t)m * D + n] = R[tn][e];
    }
  }
}

// ---------------------------------------------------------------------------
extern "C" void kernel_launch(void* const* d_in, const int* in_sizes, int n_in,
                              void* d_out, int out_size, void* d_ws,
                              size_t ws_size, hipStream_t stream) {
  const float* x0 = (const float*)d_in[0];
  const float* x1 = (const float*)d_in[1];
  const float* W0 = (const float*)d_in[2];
  const float* W1 = (const float*)d_in[3];

  float* f0 = (float*)d_out;               // (B,2,L,D)
  float* f1 = f0 + (size_t)B * 2 * L * D;  // (B,2,L,D)

  // Workspace layout (~17 MB total)
  char* ws = (char*)d_ws;
  _Float16* x0h = (_Float16*)ws; ws += (size_t)B * L * D * 2;
  _Float16* x1h = (_Float16*)ws; ws += (size_t)B * L * D * 2;
  float*    sq0 = (float*)ws;    ws += (size_t)B * L * 4;
  float*    sq1 = (float*)ws;    ws += (size_t)B * L * 4;
  _Float16* W0t = (_Float16*)ws; ws += (size_t)L * D * 2;
  _Float16* W1t = (_Float16*)ws; ws += (size_t)L * D * 2;

  prep_convert<<<(B * L * D) / 256, 256, 0, stream>>>(x0, x1, x0h, x1h, f0, f1);
  prep_sq<<<(B * L) / 8, 256, 0, stream>>>(x0, x1, sq0, sq1);
  prep_wt<<<(L * D) / 256, 256, 0, stream>>>(W0, W1, W0t, W1t);

  attn_fused<<<dim3(L / 128, B, 2), 256, 0, stream>>>(x0h, x1h, sq0, sq1, W0t,
                                                      W1t, f0, f1);
}
